// Transformer_block_85607288144147
// MI455X (gfx1250) — matrix operand order repaired and checked by reading, not compiled, not run on hardware
//
#include <hip/hip_runtime.h>
#include <cstdint>
#include <cstddef>

constexpr int TB  = 256;
constexpr int B   = 4;
constexpr int S   = 2048;
constexpr int D   = 1024;
constexpr int H   = 16;
constexpr int HD  = 64;
constexpr int DFF = 2752;
constexpr int BS  = B * S;   // 8192

typedef __attribute__((ext_vector_type(16))) __bf16 v16bf;
typedef __attribute__((ext_vector_type(8)))  __bf16 v8bf;
typedef __attribute__((ext_vector_type(8)))  float  v8f;

// ---------------- helpers ----------------

__device__ __forceinline__ __bf16 f2bf(float f) {
  unsigned u = __builtin_bit_cast(unsigned, f);
  unsigned short r = (unsigned short)((u + 0x7FFFu + ((u >> 16) & 1u)) >> 16);
  return __builtin_bit_cast(__bf16, r);
}

// A fragment (16x32 bf16, M x K): lane L: m = L%16, chunk base c0 = (L<16?0:16).
// Per-lane 16 values: cols [c0..c0+15].  Caller folds c0 in.
__device__ __forceinline__ v16bf load_afrag(const __bf16* p) {
  v8bf lo = *(const v8bf*)(p);
  v8bf hi = *(const v8bf*)(p + 8);
  v16bf r;
#pragma unroll
  for (int i = 0; i < 8; i++) { r[i] = lo[i]; r[i + 8] = hi[i]; }
  return r;
}

// B fragment (32x16 bf16, K x N): lane L: n = L%16, rows kbase+(L/16)*16 .. +15,
// contiguous 16 bf16 from a row of W[N,K] (or K/VT row). Caller folds offsets.
__device__ __forceinline__ v16bf load_bfrag(const __bf16* p) {
  return *(const v16bf*)(p);
}

__device__ __forceinline__ v8f wmma_bf16(v16bf a, v16bf b, v8f c) {
  return __builtin_amdgcn_wmma_f32_16x16x32_bf16(false, a, false, b, (short)0, c,
                                                 false, false);
}

__device__ __forceinline__ v8f zero8() {
  v8f z;
#pragma unroll
  for (int i = 0; i < 8; i++) z[i] = 0.0f;
  return z;
}

__device__ __forceinline__ float redmax16(float v) {
#pragma unroll
  for (int m = 1; m < 16; m <<= 1) v = fmaxf(v, __shfl_xor(v, m, 32));
  return v;
}
__device__ __forceinline__ float redsum16(float v) {
#pragma unroll
  for (int m = 1; m < 16; m <<= 1) v += __shfl_xor(v, m, 32);
  return v;
}

// ---------------- elementwise kernels ----------------

__global__ __launch_bounds__(TB) void convert_kernel(const float* __restrict__ in,
                                                     __bf16* __restrict__ out,
                                                     size_t n) {
  size_t i = (size_t)blockIdx.x * TB + threadIdx.x;
  if (i < n) out[i] = f2bf(in[i]);
}

__global__ __launch_bounds__(TB) void rmsnorm_kernel(const float* __restrict__ x,
                                                     const float* __restrict__ w,
                                                     __bf16* __restrict__ out) {
  __shared__ float red[TB];
  const int row = blockIdx.x;
  const float* xr = x + (size_t)row * D;
  float ss = 0.0f;
  for (int i = threadIdx.x; i < D; i += TB) { float v = xr[i]; ss += v * v; }
  red[threadIdx.x] = ss;
  __syncthreads();
  for (int s2 = TB / 2; s2 > 0; s2 >>= 1) {
    if ((int)threadIdx.x < s2) red[threadIdx.x] += red[threadIdx.x + s2];
    __syncthreads();
  }
  const float scale = rsqrtf(red[0] * (1.0f / (float)D) + 1e-5f);
  __bf16* orow = out + (size_t)row * D;
  for (int i = threadIdx.x; i < D; i += TB) orow[i] = f2bf(xr[i] * scale * w[i]);
}

// qkv fp32 [B,S,3,H,HD] -> rope(q),rope(k) bf16 [B,H,S,HD], v bf16 transposed [B,H,HD,S]
__global__ __launch_bounds__(TB) void rope_pack_kernel(const float* __restrict__ qkv,
                                                       __bf16* __restrict__ Qo,
                                                       __bf16* __restrict__ Ko,
                                                       __bf16* __restrict__ VTo) {
  int t = blockIdx.x * TB + threadIdx.x;       // < B*H*S*(HD/2)
  const int i = t & 31;                        // pair index (half = 32)
  t >>= 5;
  const int s = t & (S - 1);
  t >>= 11;
  const int h = t & (H - 1);
  const int b = t >> 4;

  const float* base = qkv + (size_t)(b * S + s) * (3 * D) + h * HD;
  const float q1 = base[2 * i],          q2 = base[2 * i + 1];
  const float k1 = base[D + 2 * i],      k2 = base[D + 2 * i + 1];
  const float v1 = base[2 * D + 2 * i],  v2 = base[2 * D + 2 * i + 1];

  // inv_freq = theta^(-2i/HD) = exp(-i * ln(1e4)/32)
  const float invf = __expf(-(float)i * (9.210340371976184f / 32.0f));
  const float ang = (float)s * invf;
  float sn, cs;
  __sincosf(ang, &sn, &cs);

  const size_t qko = (size_t)((b * H + h) * S + s) * HD + 2 * i;
  Qo[qko]     = f2bf(q1 * cs - q2 * sn);
  Qo[qko + 1] = f2bf(q1 * sn + q2 * cs);
  Ko[qko]     = f2bf(k1 * cs - k2 * sn);
  Ko[qko + 1] = f2bf(k1 * sn + k2 * cs);

  const size_t vtb = (size_t)(b * H + h) * HD * S;
  VTo[vtb + (size_t)(2 * i) * S + s]     = f2bf(v1);
  VTo[vtb + (size_t)(2 * i + 1) * S + s] = f2bf(v2);
}

__global__ __launch_bounds__(TB) void swiglu_kernel(const float* __restrict__ h1,
                                                    const float* __restrict__ h3,
                                                    __bf16* __restrict__ out,
                                                    int n) {
  int i = blockIdx.x * TB + threadIdx.x;
  if (i < n) {
    const float a = h1[i];
    const float sig = 1.0f / (1.0f + __expf(-a));
    out[i] = f2bf(a * sig * h3[i]);
  }
}

// ---------------- WMMA GEMM: C[M,N] = A[M,K] * W[N,K]^T (+ resid) ----------------
// block = 256 thr = 8 waves; block tile 128x64; wave tile 32x32.
__global__ __launch_bounds__(TB) void gemm_kernel(const __bf16* __restrict__ A,
                                                  const __bf16* __restrict__ W,
                                                  const float* __restrict__ resid,
                                                  float* __restrict__ C,
                                                  int M, int N, int K) {
  const int wave = threadIdx.x >> 5;
  const int lane = threadIdx.x & 31;
  const int wm = wave & 3;
  const int wn = wave >> 2;
  const int m0 = blockIdx.y * 128 + wm * 32;
  const int n0 = blockIdx.x * 64 + wn * 32;
  const int mrow = lane & 15;
  const int hi = lane >> 4;

  const __bf16* a0p = A + (size_t)(m0 + mrow) * K + hi * 16;
  const __bf16* a1p = a0p + (size_t)16 * K;
  const __bf16* b0p = W + (size_t)(n0 + mrow) * K + hi * 16;
  const __bf16* b1p = b0p + (size_t)16 * K;

  v8f acc[2][2];
#pragma unroll
  for (int mi = 0; mi < 2; mi++)
#pragma unroll
    for (int ni = 0; ni < 2; ni++) acc[mi][ni] = zero8();

  for (int kk = 0; kk < K; kk += 32) {
    v16bf a0 = load_afrag(a0p + kk);
    v16bf a1 = load_afrag(a1p + kk);
    v16bf b0 = load_bfrag(b0p + kk);
    v16bf b1 = load_bfrag(b1p + kk);
    acc[0][0] = wmma_bf16(a0, b0, acc[0][0]);
    acc[0][1] = wmma_bf16(a0, b1, acc[0][1]);
    acc[1][0] = wmma_bf16(a1, b0, acc[1][0]);
    acc[1][1] = wmma_bf16(a1, b1, acc[1][1]);
  }

#pragma unroll
  for (int mi = 0; mi < 2; mi++)
#pragma unroll
    for (int ni = 0; ni < 2; ni++)
#pragma unroll
      for (int r = 0; r < 8; r++) {
        const int row = m0 + mi * 16 + r + 8 * hi;
        const int col = n0 + ni * 16 + mrow;
        const size_t idx = (size_t)row * N + col;
        float v = acc[mi][ni][r];
        if (resid) v += resid[idx];
        C[idx] = v;
      }
}

// ---------------- flash attention (causal, online softmax) ----------------
// grid = (S/128, B*H); 8 waves/block, each wave owns 16 query rows.
__global__ __launch_bounds__(TB) void flash_attn_kernel(const __bf16* __restrict__ Q,
                                                        const __bf16* __restrict__ Kc,
                                                        const __bf16* __restrict__ VT,
                                                        __bf16* __restrict__ Out) {
  __shared__ __bf16 pbuf[8][16 * 32];
  const int wave = threadIdx.x >> 5;
  const int lane = threadIdx.x & 31;
  const int bh = blockIdx.y;
  const int h = bh & (H - 1);
  const int b = bh >> 4;
  const int q0 = blockIdx.x * 128 + wave * 16;
  const int mrow = lane & 15;
  const int hi = lane >> 4;

  const __bf16* qbase  = Q  + (size_t)bh * S * HD;
  const __bf16* kbase  = Kc + (size_t)bh * S * HD;
  const __bf16* vtbase = VT + (size_t)bh * HD * S;

  const __bf16* qrow = qbase + (size_t)(q0 + mrow) * HD + hi * 16;
  const v16bf aq0 = load_afrag(qrow);
  const v16bf aq1 = load_afrag(qrow + 32);

  v8f o0 = zero8(), o1 = zero8(), o2 = zero8(), o3 = zero8();
  float mR[8], lR[8];
#pragma unroll
  for (int r = 0; r < 8; r++) { mR[r] = -INFINITY; lR[r] = 0.0f; }

  const float sc = 0.125f;  // 1/sqrt(HD)
  const int kend = q0 + 15;
  for (int kb = 0; kb <= kend; kb += 32) {
    // scores: 16x32 tile = two 16x16 WMMA n-subtiles, K-dim = HD via 2 chunks
    const __bf16* kr0 = kbase + (size_t)(kb + mrow) * HD + hi * 16;
    const __bf16* kr1 = kbase + (size_t)(kb + 16 + mrow) * HD + hi * 16;
    v8f s0 = zero8(), s1 = zero8();
    s0 = wmma_bf16(aq0, load_bfrag(kr0), s0);
    s0 = wmma_bf16(aq1, load_bfrag(kr0 + 32), s0);
    s1 = wmma_bf16(aq0, load_bfrag(kr1), s1);
    s1 = wmma_bf16(aq1, load_bfrag(kr1 + 32), s1);

#pragma unroll
    for (int r = 0; r < 8; r++) {
      const int qidx = q0 + r + 8 * hi;
      float v0 = s0[r] * sc;
      float v1 = s1[r] * sc;
      if (kb + mrow > qidx) v0 = -INFINITY;
      if (kb + 16 + mrow > qidx) v1 = -INFINITY;
      const float mx = redmax16(fmaxf(v0, v1));
      const float mnew = fmaxf(mR[r], mx);
      const float corr = __expf(mR[r] - mnew);
      const float p0 = __expf(v0 - mnew);
      const float p1 = __expf(v1 - mnew);
      const float rs = redsum16(p0 + p1);
      lR[r] = lR[r] * corr + rs;
      mR[r] = mnew;
      o0[r] *= corr; o1[r] *= corr; o2[r] *= corr; o3[r] *= corr;
      const int prow = r + 8 * hi;
      pbuf[wave][prow * 32 + mrow]      = f2bf(p0);
      pbuf[wave][prow * 32 + 16 + mrow] = f2bf(p1);
    }

    // intra-wave cross-lane LDS handoff: wait for DS stores before re-lane load
    asm volatile("s_wait_dscnt 0" ::: "memory");

    v16bf pf;
    const int pbase = mrow * 32 + hi * 16;
#pragma unroll
    for (int i = 0; i < 8; i++) {
      pf[i]     = pbuf[wave][pbase + i];
      pf[i + 8] = pbuf[wave][pbase + 8 + i];
    }

    // O += P(16x32) @ V(32x64): 4 n-tiles from transposed V
    const __bf16* vr = vtbase + (size_t)mrow * S + kb + hi * 16;
    o0 = wmma_bf16(pf, load_bfrag(vr), o0);
    o1 = wmma_bf16(pf, load_bfrag(vr + (size_t)16 * S), o1);
    o2 = wmma_bf16(pf, load_bfrag(vr + (size_t)32 * S), o2);
    o3 = wmma_bf16(pf, load_bfrag(vr + (size_t)48 * S), o3);
  }

#pragma unroll
  for (int r = 0; r < 8; r++) {
    const float inv = 1.0f / lR[r];
    const int row = q0 + r + 8 * hi;
    __bf16* orow = Out + (size_t)(b * S + row) * D + h * HD;
    orow[mrow]      = f2bf(o0[r] * inv);
    orow[16 + mrow] = f2bf(o1[r] * inv);
    orow[32 + mrow] = f2bf(o2[r] * inv);
    orow[48 + mrow] = f2bf(o3[r] * inv);
  }
}

// ---------------- launch ----------------

extern "C" void kernel_launch(void* const* d_in, const int* in_sizes, int n_in,
                              void* d_out, int out_size, void* d_ws, size_t ws_size,
                              hipStream_t stream) {
  (void)in_sizes; (void)n_in; (void)out_size; (void)ws_size;

  const float* x     = (const float*)d_in[0];
  const float* n1w   = (const float*)d_in[1];
  const float* n2w   = (const float*)d_in[2];
  const float* qkv_w = (const float*)d_in[3];
  const float* o_w   = (const float*)d_in[4];
  const float* w1    = (const float*)d_in[5];
  const float* w3    = (const float*)d_in[6];
  const float* w2    = (const float*)d_in[7];
  float* out = (float*)d_out;

  char* p = (char*)d_ws;
  auto carve = [&](size_t bytes) -> void* {
    void* r = (void*)p;
    p += (bytes + 255) & ~(size_t)255;
    return r;
  };
  __bf16* wqkv_b = (__bf16*)carve((size_t)3 * D * D * 2);
  __bf16* wo_b   = (__bf16*)carve((size_t)D * D * 2);
  __bf16* w1_b   = (__bf16*)carve((size_t)DFF * D * 2);
  __bf16* w3_b   = (__bf16*)carve((size_t)DFF * D * 2);
  __bf16* w2_b   = (__bf16*)carve((size_t)D * DFF * 2);
  __bf16* xn_b   = (__bf16*)carve((size_t)BS * D * 2);
  __bf16* q_b    = (__bf16*)carve((size_t)BS * D * 2);
  __bf16* k_b    = (__bf16*)carve((size_t)BS * D * 2);
  __bf16* vt_b   = (__bf16*)carve((size_t)BS * D * 2);
  __bf16* attn_b = (__bf16*)carve((size_t)BS * D * 2);
  __bf16* hg_b   = (__bf16*)carve((size_t)BS * DFF * 2);
  float*  x1     = (float*)carve((size_t)BS * D * 4);
  float*  qkv_f  = (float*)carve((size_t)BS * 3 * D * 4);
  float*  h1_f   = qkv_f;  // reuse: qkv fully consumed by rope before FFN GEMMs
  float*  h3_f   = (float*)carve((size_t)BS * DFF * 4);

  auto cvt = [&](const float* src, __bf16* dst, size_t n) {
    convert_kernel<<<(unsigned)((n + TB - 1) / TB), TB, 0, stream>>>(src, dst, n);
  };
  cvt(qkv_w, wqkv_b, (size_t)3 * D * D);
  cvt(o_w,   wo_b,   (size_t)D * D);
  cvt(w1,    w1_b,   (size_t)DFF * D);
  cvt(w3,    w3_b,   (size_t)DFF * D);
  cvt(w2,    w2_b,   (size_t)D * DFF);

  // x -> rmsnorm -> bf16
  rmsnorm_kernel<<<BS, TB, 0, stream>>>(x, n1w, xn_b);
  // qkv = xn @ Wqkv^T  : [8192,3072]
  gemm_kernel<<<dim3(3 * D / 64, BS / 128), TB, 0, stream>>>(xn_b, wqkv_b, nullptr,
                                                             qkv_f, BS, 3 * D, D);
  // rope + layout pack (q,k [B,H,S,HD] bf16; v transposed [B,H,HD,S] bf16)
  rope_pack_kernel<<<(B * H * S * (HD / 2)) / TB, TB, 0, stream>>>(qkv_f, q_b, k_b, vt_b);
  // causal flash attention -> attn_b [B,S,D] bf16
  flash_attn_kernel<<<dim3(S / 128, B * H), TB, 0, stream>>>(q_b, k_b, vt_b, attn_b);
  // x1 = x + attn @ Wo^T
  gemm_kernel<<<dim3(D / 64, BS / 128), TB, 0, stream>>>(attn_b, wo_b, x, x1, BS, D, D);
  // x1 -> rmsnorm -> bf16 (reuse xn_b)
  rmsnorm_kernel<<<BS, TB, 0, stream>>>(x1, n2w, xn_b);
  // h1 = xn @ W1^T ; h3 = xn @ W3^T : [8192,2752]
  gemm_kernel<<<dim3(DFF / 64, BS / 128), TB, 0, stream>>>(xn_b, w1_b, nullptr, h1_f,
                                                           BS, DFF, D);
  gemm_kernel<<<dim3(DFF / 64, BS / 128), TB, 0, stream>>>(xn_b, w3_b, nullptr, h3_f,
                                                           BS, DFF, D);
  // hg = silu(h1)*h3 -> bf16
  swiglu_kernel<<<(BS * DFF) / TB, TB, 0, stream>>>(h1_f, h3_f, hg_b, BS * DFF);
  // out = x1 + hg @ W2^T
  gemm_kernel<<<dim3(D / 64, BS / 128), TB, 0, stream>>>(hg_b, w2_b, x1, out, BS, D, DFF);
}
